// DCNv4_13331578486941
// MI455X (gfx1250) — compile-verified, matrix operand order checked by reading, not run on hardware
//
#include <hip/hip_runtime.h>
#include <hip/hip_bf16.h>
#include <math.h>

// ---------------------------------------------------------------------------
// DCNv4 for MI455X (gfx1250, wave32).
//   x --f16--> [WMMA proj v] [WMMA proj offset/mask] --> bilinear gather core
//     --> [WMMA output proj] -> out (f32)
// GEMM: block = 8 waves, block tile 128(M) x 64(N); the 64x256 f16 weight tile
// is loop-invariant and staged once into LDS (padded stride, conflict-free
// ds_load_b128); each wave computes a 16x64 strip = 4 WMMA accumulators, so one
// A fragment feeds 4 v_wmma_f32_16x16x32_f16.
// ---------------------------------------------------------------------------

typedef _Float16 v16h __attribute__((ext_vector_type(16)));
typedef _Float16 v8h  __attribute__((ext_vector_type(8)));
typedef float    v8f  __attribute__((ext_vector_type(8)));
typedef float    v4f  __attribute__((ext_vector_type(4)));

#define N_IMG   4
#define H_DIM   64
#define W_DIM   64
#define L_DIM   (H_DIM * W_DIM)        // 4096
#define C_DIM   256
#define G_DIM   8
#define CG_DIM  32
#define K2_DIM  9
#define OM_DIM  216                    // G * K2 * 3
#define M_ROWS  (N_IMG * L_DIM)        // 16384

#define BN          64                 // block column tile
#define LDS_STRIDE  264                // 256 + 8 halves pad: 528B = 132 dwords -> lane n hits bank 4n

// ----------------------------- f32 -> f16 (x8 vectorized) -------------------
__global__ void f32_to_f16_vec8(const float* __restrict__ src,
                                _Float16* __restrict__ dst, int n8) {
  int i = blockIdx.x * blockDim.x + threadIdx.x;
  if (i < n8) {
    v4f lo = ((const v4f*)src)[2 * i];
    v4f hi = ((const v4f*)src)[2 * i + 1];
    v8h o;
    #pragma unroll
    for (int j = 0; j < 4; j++) {
      o[j]     = (_Float16)lo[j];
      o[4 + j] = (_Float16)hi[j];
    }
    ((v8h*)dst)[i] = o;
  }
}

// --------------------------- WMMA GEMM (A * W^T + b) ------------------------
// D[m][n] = sum_k A[m][k] * W[n][k] + bias[n], K = 256 fixed.
// A: (M x 256) f16 row-major, W: (Nout x 256) f16 row-major, D: f32 (M x Nout).
// grid = (ceil(Nout/64), M/128), block = 256 threads (8 waves).
__global__ void wmma_gemm_nt_bias(const _Float16* __restrict__ A,
                                  const _Float16* __restrict__ W,
                                  const float* __restrict__ bias,
                                  float* __restrict__ D,
                                  int Nout) {
  __shared__ __align__(16) _Float16 Bs[BN * LDS_STRIDE];   // ~33 KB of 320 KB/WGP

  const int tid    = threadIdx.x;
  const int lane   = tid & 31;
  const int wave   = tid >> 5;
  const int n0     = blockIdx.x * BN;
  const int tile_m = blockIdx.y * 8 + wave;

  // ---- Stage the 64x256 weight tile into LDS once (it is k-loop invariant).
  {
    const int j  = tid >> 2;          // row within tile: 0..63
    const int ks = (tid & 3) * 64;    // K chunk start
    const int nrow   = n0 + j;
    const int nclamp = nrow < Nout ? nrow : (Nout - 1);   // clamp, no divergence
    const _Float16* __restrict__ src = W + (size_t)nclamp * C_DIM + ks;
    _Float16* dst = &Bs[j * LDS_STRIDE + ks];
    #pragma unroll
    for (int i = 0; i < 64; i += 8)
      *(v8h*)(dst + i) = *(const v8h*)(src + i);
  }
  __syncthreads();

  // ISA 7.12.2 A/B fragment layout: lane holds K = {kb..kb+7, 16+kb..16+kb+7},
  // kb = 8 for lanes 16-31.
  const int r        = lane & 15;
  const int half_sel = lane >> 4;
  const int kbase    = half_sel * 8;

  const _Float16* __restrict__ arow = A + (size_t)(tile_m * 16 + r) * C_DIM;

  v8f acc[4] = {{}, {}, {}, {}};
  #pragma unroll
  for (int kk = 0; kk < C_DIM; kk += 32) {
    v8h a_lo = *(const v8h*)(arow + kk + kbase);
    v8h a_hi = *(const v8h*)(arow + kk + 16 + kbase);
    v16h a = __builtin_shufflevector(a_lo, a_hi,
                                     0,1,2,3,4,5,6,7,8,9,10,11,12,13,14,15);
    #pragma unroll
    for (int t = 0; t < 4; t++) {
      const _Float16* brow = &Bs[(t * 16 + r) * LDS_STRIDE + kk + kbase];
      v8h b_lo = *(const v8h*)(brow);
      v8h b_hi = *(const v8h*)(brow + 16);
      v16h b = __builtin_shufflevector(b_lo, b_hi,
                                       0,1,2,3,4,5,6,7,8,9,10,11,12,13,14,15);
      acc[t] = __builtin_amdgcn_wmma_f32_16x16x32_f16(
          /*neg_a=*/false, a, /*neg_b=*/false, b,
          /*c_mod=*/(short)0, acc[t], /*reuse_a=*/false, /*reuse_b=*/false);
    }
  }

  // C/D layout: lane = column (mod 16), VGPR rr = row rr + 8*(lane>=16).
  const int ncol_lo = n0 + (lane & 15);
  const int mbase   = tile_m * 16 + 8 * half_sel;
  #pragma unroll
  for (int t = 0; t < 4; t++) {
    const int ncol = ncol_lo + t * 16;
    if (ncol < Nout) {
      const float bv = bias[ncol];
      #pragma unroll
      for (int rr = 0; rr < 8; rr++)
        D[(size_t)(mbase + rr) * Nout + ncol] = acc[t][rr] + bv;
    }
  }
}

// ------------------------------ DCNv4 core ----------------------------------
// One wave per (n, l, g); lane = channel within group (Cg == 32 == wave32).
// om layout per (n,l): [g*27 + 2k] = off_w, [g*27 + 2k + 1] = off_h,
//                      [g*27 + 18 + k] = mask.  STRIDE=DIL=OFFSET_SCALE=1, PAD=c=1.
__global__ void dcnv4_core_kernel(const float* __restrict__ v,
                                  const float* __restrict__ om,
                                  _Float16* __restrict__ o_h) {
  const int wave = blockIdx.x * (blockDim.x >> 5) + (threadIdx.x >> 5);
  const int lane = threadIdx.x & 31;

  const int g = wave & (G_DIM - 1);
  const int l = (wave >> 3) & (L_DIM - 1);
  const int n = wave >> 15;
  const int hh = l >> 6;
  const int ww = l & (W_DIM - 1);

  const float* __restrict__ omp = om + ((size_t)n * L_DIM + l) * OM_DIM + g * 27;
  const float* __restrict__ vg  = v  + (size_t)n * L_DIM * C_DIM + g * CG_DIM + lane;

  float acc = 0.0f;
  #pragma unroll
  for (int k = 0; k < K2_DIM; k++) {
    const float ow = omp[2 * k];        // same addr across lanes -> broadcast
    const float oh = omp[2 * k + 1];
    const float mk = omp[18 + k];

    const float loc_h = (float)hh + (float)(k % 3) - 1.0f + oh;
    const float loc_w = (float)ww + (float)(k / 3) - 1.0f + ow;
    const float h0f = floorf(loc_h);
    const float w0f = floorf(loc_w);
    const float fh = loc_h - h0f;
    const float fw = loc_w - w0f;
    const int h0 = (int)h0f;
    const int w0 = (int)w0f;

    #pragma unroll
    for (int dh = 0; dh < 2; dh++) {
      #pragma unroll
      for (int dw = 0; dw < 2; dw++) {
        const int hi = h0 + dh;
        const int wi = w0 + dw;
        const bool valid = (hi >= 0) & (hi < H_DIM) & (wi >= 0) & (wi < W_DIM);
        const float wgt = (dh ? fh : 1.0f - fh) * (dw ? fw : 1.0f - fw);
        const int hic = hi < 0 ? 0 : (hi > H_DIM - 1 ? H_DIM - 1 : hi);
        const int wic = wi < 0 ? 0 : (wi > W_DIM - 1 ? W_DIM - 1 : wi);
        const int idx = hic * W_DIM + wic;
        // 32 lanes read 32 consecutive floats: one coalesced 128B row.
        const float samp = vg[(size_t)idx * C_DIM];
        acc += valid ? (mk * wgt * samp) : 0.0f;
      }
    }
  }

  o_h[((size_t)n * L_DIM + l) * C_DIM + g * CG_DIM + lane] = (_Float16)acc;
}

// ------------------------------- launch -------------------------------------
extern "C" void kernel_launch(void* const* d_in, const int* in_sizes, int n_in,
                              void* d_out, int out_size, void* d_ws, size_t ws_size,
                              hipStream_t stream) {
  const float* x    = (const float*)d_in[0];   // (4, 4096, 256)
  const float* Wv   = (const float*)d_in[1];   // (256, 256)
  const float* bv   = (const float*)d_in[2];   // (256,)
  const float* Wom  = (const float*)d_in[3];   // (216, 256)
  const float* bom  = (const float*)d_in[4];   // (216,)
  const float* Wo   = (const float*)d_in[5];   // (256, 256)
  const float* bo   = (const float*)d_in[6];   // (256,)
  float* out = (float*)d_out;                  // (4, 4096, 256) f32

  // Workspace carve-out (256B aligned regions).
  char* ws = (char*)d_ws;
  size_t off = 0;
  auto carve = [&](size_t bytes) -> char* {
    char* p = ws + off;
    off = (off + bytes + 255) & ~(size_t)255;
    return p;
  };
  _Float16* x_h   = (_Float16*)carve((size_t)M_ROWS * C_DIM * 2);   // 8 MB
  _Float16* Wv_h  = (_Float16*)carve((size_t)C_DIM * C_DIM * 2);
  _Float16* Wom_h = (_Float16*)carve((size_t)OM_DIM * C_DIM * 2);
  _Float16* Wo_h  = (_Float16*)carve((size_t)C_DIM * C_DIM * 2);
  float*    v_f   = (float*)   carve((size_t)M_ROWS * C_DIM * 4);   // 16 MB
  float*    om_f  = (float*)   carve((size_t)M_ROWS * OM_DIM * 4);  // 14 MB
  _Float16* o_h   = x_h;   // x_h is dead after the two projection GEMMs; reuse.

  // 1) Stage inputs/weights to f16 (8 floats per thread).
  {
    const int n1 = (M_ROWS * C_DIM) / 8;
    f32_to_f16_vec8<<<(n1 + 255) / 256, 256, 0, stream>>>(x, x_h, n1);
    const int n2 = (C_DIM * C_DIM) / 8;
    f32_to_f16_vec8<<<(n2 + 255) / 256, 256, 0, stream>>>(Wv, Wv_h, n2);
    const int n3 = (OM_DIM * C_DIM) / 8;
    f32_to_f16_vec8<<<(n3 + 255) / 256, 256, 0, stream>>>(Wom, Wom_h, n3);
    f32_to_f16_vec8<<<(n2 + 255) / 256, 256, 0, stream>>>(Wo, Wo_h, n2);
  }

  // 2) Projections via WMMA: grid = (ceil(Nout/64), M/128), block = 8 waves.
  {
    dim3 blk(256);
    dim3 grid_v((C_DIM + BN - 1) / BN, M_ROWS / 128);    // (4, 128)
    wmma_gemm_nt_bias<<<grid_v, blk, 0, stream>>>(x_h, Wv_h, bv, v_f, C_DIM);
    dim3 grid_om((OM_DIM + BN - 1) / BN, M_ROWS / 128);  // (4, 128)
    wmma_gemm_nt_bias<<<grid_om, blk, 0, stream>>>(x_h, Wom_h, bom, om_f, OM_DIM);
  }

  // 3) Deformable gather core: N*L*G waves, 8 waves per block.
  {
    const int total_waves = N_IMG * L_DIM * G_DIM;   // 131072
    dcnv4_core_kernel<<<total_waves / 8, 256, 0, stream>>>(v_f, om_f, o_h);
  }

  // 4) Output projection straight into d_out (f32).
  {
    dim3 blk(256);
    dim3 grid_o((C_DIM + BN - 1) / BN, M_ROWS / 128);
    wmma_gemm_nt_bias<<<grid_o, blk, 0, stream>>>(o_h, Wo_h, bo, out, C_DIM);
  }
}
